// RandomSynthesizedMultiHeadSelfAttention_37048387895403
// MI455X (gfx1250) — compile-verified, matrix operand order checked by reading, not run on hardware
//
#include <hip/hip_runtime.h>

typedef _Float16 v16h __attribute__((ext_vector_type(16)));
typedef _Float16 v8h  __attribute__((ext_vector_type(8)));
typedef _Float16 v4h  __attribute__((ext_vector_type(4)));
typedef float    v8f  __attribute__((ext_vector_type(8)));

#define BATCH   4
#define SEQ     1024
#define DIM     1024
#define NH      16
#define HD      64
#define MAXREL  8

// ---------------------------------------------------------------------------
// WMMA fragment loader (f16 source). CDNA5 16-bit A/B layout (ISA 7.12.2):
//   lanes 0-15 : row = lane,    elems 0..7 -> K=k0+0..7,  elems 8..15 -> K=k0+16..23
//   lanes16-31 : row = lane-16, elems 0..7 -> K=k0+8..15, elems 8..15 -> K=k0+24..31
// Both A and B use "row over K per lane" (for B, row == output column n since
// we always multiply by W^T / pre-transposed V).
// ---------------------------------------------------------------------------
__device__ __forceinline__ v16h frag_from_f16(const _Float16* __restrict__ base,
                                              int row, int ld, int k0, int lane) {
  const _Float16* p = base + (size_t)row * ld + k0 + ((lane >> 4) << 3);
  v8h lo = *(const v8h*)(p);        // K = kbase + 0..7
  v8h hi = *(const v8h*)(p + 16);   // K = kbase + 16..23
  return __builtin_shufflevector(lo, hi, 0,1,2,3,4,5,6,7,8,9,10,11,12,13,14,15);
}

#define WMMA(A, B, C) __builtin_amdgcn_wmma_f32_16x16x32_f16(false, (A), false, (B), (short)0, (C), false, false)

// ---------------------------------------------------------------------------
// Double-buffered 16(M) x 64(N) x 1024(K) strip GEMM: next k-step's fragments
// (A + 4xB, ten b128 loads) are issued BEFORE the current step's 4 WMMAs, so
// each WMMA waits only on loads launched one iteration earlier.
// ---------------------------------------------------------------------------
__device__ __forceinline__ void gemm_strip(const _Float16* __restrict__ A, int arow, int lda,
                                           const _Float16* __restrict__ B, int brow, int ldb,
                                           int lane, v8f c[4]) {
  v16h a  = frag_from_f16(A, arow,      lda, 0, lane);
  v16h b0 = frag_from_f16(B, brow,      ldb, 0, lane);
  v16h b1 = frag_from_f16(B, brow + 16, ldb, 0, lane);
  v16h b2 = frag_from_f16(B, brow + 32, ldb, 0, lane);
  v16h b3 = frag_from_f16(B, brow + 48, ldb, 0, lane);
  for (int k0 = 32; k0 < 1024; k0 += 32) {
    v16h an  = frag_from_f16(A, arow,      lda, k0, lane);
    v16h bn0 = frag_from_f16(B, brow,      ldb, k0, lane);
    v16h bn1 = frag_from_f16(B, brow + 16, ldb, k0, lane);
    v16h bn2 = frag_from_f16(B, brow + 32, ldb, k0, lane);
    v16h bn3 = frag_from_f16(B, brow + 48, ldb, k0, lane);
    c[0] = WMMA(a, b0, c[0]);
    c[1] = WMMA(a, b1, c[1]);
    c[2] = WMMA(a, b2, c[2]);
    c[3] = WMMA(a, b3, c[3]);
    a = an; b0 = bn0; b1 = bn1; b2 = bn2; b3 = bn3;
  }
  c[0] = WMMA(a, b0, c[0]);
  c[1] = WMMA(a, b1, c[1]);
  c[2] = WMMA(a, b2, c[2]);
  c[3] = WMMA(a, b3, c[3]);
}

// ---------------------------------------------------------------------------
// Kernel 0: one-shot f32 -> f16 conversion (x, W_in, W_out). Each thread
// converts 8 elements: two b128 loads -> packed cvts -> one b128 store.
// ---------------------------------------------------------------------------
__global__ void __launch_bounds__(256)
k_cvt_f16(const float* __restrict__ src, _Float16* __restrict__ dst) {
  const size_t i = ((size_t)blockIdx.x * 256 + threadIdx.x) * 8;
  float4 a = *(const float4*)(src + i);
  float4 b = *(const float4*)(src + i + 4);
  v8h h;
  h[0]=(_Float16)a.x; h[1]=(_Float16)a.y; h[2]=(_Float16)a.z; h[3]=(_Float16)a.w;
  h[4]=(_Float16)b.x; h[5]=(_Float16)b.y; h[6]=(_Float16)b.z; h[7]=(_Float16)b.w;
  *(v8h*)(dst + i) = h;
}

// ---------------------------------------------------------------------------
// Kernel 1: value = x @ W_in^T (f16 in, f32 accum), written TRANSPOSED as
// f16 Vt[b][h][d][j]. Wave strip = 16 rows (j) x 64 cols (h*64+d).
// 512 blocks x 8 waves = 4096 waves. All fragment loads are 16B contiguous.
// ---------------------------------------------------------------------------
__global__ void __launch_bounds__(256)
k_gemm_value(const _Float16* __restrict__ xh, const _Float16* __restrict__ Win,
             _Float16* __restrict__ Vt) {
  const int lane  = threadIdx.x & 31;
  const int wave  = threadIdx.x >> 5;
  const int strip = blockIdx.x * 8 + wave;     // 0..4095
  const int m0    = (strip >> 4) * 16;         // global row (b*1024 + j)
  const int n0    = (strip & 15) * 64;         // output col base (h*64+d)

  v8f c[4] = {};
  gemm_strip(xh, m0 + (lane & 15), DIM, Win, n0 + (lane & 15), DIM, lane, c);

  // C layout: lane l holds N = tile_n + (l&15), M-rows v + 8*(l>>4)  (v = 0..7)
  const int b     = m0 >> 10;
  const int jbase = (m0 & 1023) + ((lane >> 4) << 3);   // 8 consecutive j per lane
#pragma unroll
  for (int t = 0; t < 4; ++t) {
    const int n = n0 + t * 16 + (lane & 15);
    const int h = n >> 6, d = n & 63;
    v8h hv;
#pragma unroll
    for (int v = 0; v < 8; ++v) hv[v] = (_Float16)c[t][v];
    *(v8h*)(Vt + (((size_t)(b * NH + h) * HD + d) * SEQ + jbase)) = hv;
  }
}

// ---------------------------------------------------------------------------
// Kernel 2: fused softmax + bucketed relative-position reduction.
// One block per (h,s) row. probs -> f16; rel_out[h][s][d] = sum_k pb[k]*rel[k][d]
// where pb[k] are the 17 clip-bucket probability sums (bucket 0: j<=s-8,
// bucket 16: j>=s+8, buckets 1..15 are single elements j=s+k-8).
// ---------------------------------------------------------------------------
__global__ void __launch_bounds__(256)
k_softmax_rel(const float* __restrict__ scores, const float* __restrict__ relv,
              _Float16* __restrict__ probs, float* __restrict__ relout) {
  __shared__ float2 red[256];
  __shared__ float  pb[17];
  const int tid = threadIdx.x;
  const int h   = blockIdx.x >> 10;
  const int s   = blockIdx.x & 1023;
  if (tid < 17) pb[tid] = 0.0f;

  const float* row = scores + ((size_t)h * SEQ + s) * SEQ;
  float4 v = *(const float4*)(row + tid * 4);

  // row max
  red[tid].x = fmaxf(fmaxf(v.x, v.y), fmaxf(v.z, v.w));
  __syncthreads();
  for (int off = 128; off > 0; off >>= 1) {
    if (tid < off) red[tid].x = fmaxf(red[tid].x, red[tid + off].x);
    __syncthreads();
  }
  const float rmax = red[0].x;
  __syncthreads();

  float e0 = __expf(v.x - rmax), e1 = __expf(v.y - rmax);
  float e2 = __expf(v.z - rmax), e3 = __expf(v.w - rmax);
  red[tid].x = e0 + e1 + e2 + e3;
  __syncthreads();
  for (int off = 128; off > 0; off >>= 1) {
    if (tid < off) red[tid].x += red[tid + off].x;
    __syncthreads();
  }
  const float inv = 1.0f / red[0].x;
  __syncthreads();

  float p[4] = {e0 * inv, e1 * inv, e2 * inv, e3 * inv};
  v4h ph; ph[0]=(_Float16)p[0]; ph[1]=(_Float16)p[1]; ph[2]=(_Float16)p[2]; ph[3]=(_Float16)p[3];
  *(v4h*)(probs + ((size_t)h * SEQ + s) * SEQ + tid * 4) = ph;

  // bucket probabilities: idx = clip(j - s, -8, 8) + 8
  float lo = 0.0f, hi = 0.0f;
#pragma unroll
  for (int i = 0; i < 4; ++i) {
    const int d = tid * 4 + i - s;
    if (d <= -MAXREL)      lo += p[i];
    else if (d >= MAXREL)  hi += p[i];
    else                   pb[d + MAXREL] = p[i];   // unique writer per bucket
  }
  red[tid] = make_float2(lo, hi);
  __syncthreads();
  for (int off = 128; off > 0; off >>= 1) {
    if (tid < off) { red[tid].x += red[tid + off].x; red[tid].y += red[tid + off].y; }
    __syncthreads();
  }
  if (tid == 0) { pb[0] = red[0].x; pb[16] = red[0].y; }
  __syncthreads();

  if (tid < HD) {
    float acc = 0.0f;
#pragma unroll
    for (int k = 0; k < 17; ++k) acc += pb[k] * relv[k * HD + tid];
    relout[((size_t)h * SEQ + s) * HD + tid] = acc;
  }
}

// ---------------------------------------------------------------------------
// Kernel 3: out1[b,h,s,d] = sum_j probs[h,s,j] * Vt[b,h,d,j]  (+ rel_out[h,s,d])
// written as f16 attn[b][s][h*64+d]. Wave strip = 16 s-rows x full 64 d.
// probs rows are shared across the 4 batches -> L2 reuse.
// ---------------------------------------------------------------------------
__global__ void __launch_bounds__(256)
k_attn(const _Float16* __restrict__ probs, const _Float16* __restrict__ Vt,
       const float* __restrict__ relout, _Float16* __restrict__ attn) {
  const int lane  = threadIdx.x & 31;
  const int wave  = threadIdx.x >> 5;
  const int strip = blockIdx.x * 8 + wave;   // 0..4095
  const int s0    = (strip & 63) * 16;
  const int h     = (strip >> 6) & 15;
  const int b     = strip >> 10;

  const _Float16* prow  = probs + (size_t)h * SEQ * SEQ;
  const _Float16* vbase = Vt + (size_t)(b * NH + h) * HD * SEQ;

  v8f c[4] = {};
  gemm_strip(prow, s0 + (lane & 15), SEQ, vbase, lane & 15, SEQ, lane, c);

#pragma unroll
  for (int t = 0; t < 4; ++t) {
    const int d = t * 16 + (lane & 15);
#pragma unroll
    for (int v = 0; v < 8; ++v) {
      const int s = s0 + ((lane >> 4) << 3) + v;
      const float val = c[t][v] + relout[((size_t)h * SEQ + s) * HD + d];
      attn[((size_t)(b * SEQ + s)) * DIM + h * HD + d] = (_Float16)val;
    }
  }
}

// ---------------------------------------------------------------------------
// Kernel 4: d_out = attn @ W_out^T  (f16 A, pre-converted f16 weights, f32 out)
// ---------------------------------------------------------------------------
__global__ void __launch_bounds__(256)
k_gemm_out(const _Float16* __restrict__ A, const _Float16* __restrict__ Wout,
           float* __restrict__ out) {
  const int lane  = threadIdx.x & 31;
  const int wave  = threadIdx.x >> 5;
  const int strip = blockIdx.x * 8 + wave;   // 0..4095
  const int m0    = (strip >> 4) * 16;
  const int n0    = (strip & 15) * 64;

  v8f c[4] = {};
  gemm_strip(A, m0 + (lane & 15), DIM, Wout, n0 + (lane & 15), DIM, lane, c);

#pragma unroll
  for (int t = 0; t < 4; ++t) {
    const int n = n0 + t * 16 + (lane & 15);
#pragma unroll
    for (int v = 0; v < 8; ++v) {
      const int m = m0 + ((lane >> 4) << 3) + v;
      out[(size_t)m * DIM + n] = c[t][v];
    }
  }
}

// ---------------------------------------------------------------------------
extern "C" void kernel_launch(void* const* d_in, const int* in_sizes, int n_in,
                              void* d_out, int out_size, void* d_ws, size_t ws_size,
                              hipStream_t stream) {
  (void)in_sizes; (void)n_in; (void)out_size; (void)ws_size;
  const float* x      = (const float*)d_in[0];   // (4,1024,1024)
  const float* Win    = (const float*)d_in[1];   // (1024,1024)
  const float* Wout   = (const float*)d_in[2];   // (1024,1024)
  const float* scores = (const float*)d_in[3];   // (1,16,1024,1024)
  const float* relv   = (const float*)d_in[4];   // (17,64)
  float* out = (float*)d_out;

  char* ws = (char*)d_ws;
  _Float16* Vt     = (_Float16*)(ws);                         //  8 MB  [b][h][d][j]
  _Float16* probs  = (_Float16*)(ws + ((size_t)8  << 20));    // 32 MB  [h][s][j]
  float*    relout = (float*)   (ws + ((size_t)40 << 20));    //  4 MB  [h][s][d]
  _Float16* attn   = (_Float16*)(ws + ((size_t)44 << 20));    //  8 MB  [b][s][D]
  _Float16* xh     = (_Float16*)(ws + ((size_t)52 << 20));    //  8 MB  [b*s][D]
  _Float16* Winh   = (_Float16*)(ws + ((size_t)60 << 20));    //  2 MB
  _Float16* Wouth  = (_Float16*)(ws + ((size_t)62 << 20));    //  2 MB

  // one-shot f32->f16 staging (8 elems/thread)
  k_cvt_f16<<<(BATCH * SEQ * DIM) / (256 * 8), 256, 0, stream>>>(x, xh);
  k_cvt_f16<<<(DIM * DIM) / (256 * 8), 256, 0, stream>>>(Win, Winh);
  k_cvt_f16<<<(DIM * DIM) / (256 * 8), 256, 0, stream>>>(Wout, Wouth);

  k_gemm_value<<<512, 256, 0, stream>>>(xh, Winh, Vt);
  k_softmax_rel<<<NH * SEQ, 256, 0, stream>>>(scores, relv, probs, relout);
  k_attn<<<512, 256, 0, stream>>>(probs, Vt, relout, attn);
  k_gemm_out<<<512, 256, 0, stream>>>(attn, Wouth, out);
}
